// ProcessGNN_33749853012400
// MI455X (gfx1250) — matrix operand order repaired and checked by reading, not compiled, not run on hardware
//
#include <hip/hip_runtime.h>
#include <hip/hip_fp16.h>

typedef __attribute__((ext_vector_type(16))) _Float16 v16h;
typedef __attribute__((ext_vector_type(8)))  _Float16 h8;
typedef __attribute__((ext_vector_type(8)))  float    v8f;

#define NNODES  50000
#define NEDGES  600000
#define NGRAPHS 256

// ---------- helpers ----------

// Order-preserving f32 -> u32 encoding so atomicMax(u32) == float max.
__device__ __forceinline__ unsigned encMax(float f) {
  unsigned u = __float_as_uint(f);
  return (u & 0x80000000u) ? ~u : (u | 0x80000000u);
}
__device__ __forceinline__ float decMax(unsigned u) {
  return __uint_as_float((u & 0x80000000u) ? (u ^ 0x80000000u) : ~u);
}
// HW float atomic add (global_atomic_add_f32)
__device__ __forceinline__ void atomAddF(float* p, float v) {
  unsafeAtomicAdd(p, v);
}

// 8-element A-fragment loads, forced to 128-bit vector accesses.
__device__ __forceinline__ h8 loadA8(const _Float16* p) {
  return *(const h8*)p;
}
__device__ __forceinline__ h8 loadA8(const float* p) {
  v8f f = *(const v8f*)p;   // 2 x b128
  h8 r;
#pragma unroll
  for (int i = 0; i < 8; ++i) r[i] = (_Float16)f[i];
  return r;
}

// ---------- utility kernels ----------

__global__ __launch_bounds__(256) void fill_u32_kernel(unsigned* p, unsigned v, int n) {
  int i = blockIdx.x * blockDim.x + threadIdx.x;
  if (i < n) p[i] = v;
}

// Pack B[KxN] f32 row-major into per-(colTile,kStep,lane) WMMA fragment order:
// Bp[(((ct*ksteps)+s)*32 + lane)*16 + e] = (f16) B[(32s + (lane>>4)*16 + e)*N + 16ct + (lane&15)]
// so each lane's 16 halves are contiguous (two b128 loads in the GEMM).
__global__ __launch_bounds__(256) void pack_b_kernel(const float* __restrict__ B,
                                                     _Float16* __restrict__ Bp,
                                                     int K, int N) {
  int i = blockIdx.x * blockDim.x + threadIdx.x;
  if (i >= K * N) return;
  const int e    = i & 15;
  const int l    = (i >> 4) & 31;
  const int rest = i >> 9;            // ct*ksteps + s
  const int ksteps = K >> 5;
  const int ct = rest / ksteps;
  const int s  = rest % ksteps;
  const int k  = (s << 5) + ((l >> 4) << 4) + e;
  const int n  = (ct << 4) + (l & 15);
  Bp[i] = (_Float16)B[(size_t)k * N + n];
}

// ---------- WMMA GEMM: D(f16) = act(A[MxK] * B[KxN] + bias) ----------
// One wave computes a 16x64 tile: one A fragment feeds 4 WMMAs per K-step.
// All 4 B fragments (8 x b128) are issued before the WMMA sequence so the
// matrix pipe overlaps with in-flight loads (partial s_wait_loadcnt).
// 8 waves/block share one column group (B fragments reused out of L0/L2).
// A/B fragments use contiguous k = 16*(lane/16) + s (wave32). Requires M%16==0, N%64==0, K%32==0.
template <typename AT, bool RELU>
__global__ __launch_bounds__(256) void wmma_gemm_kernel(
    const AT* __restrict__ A, const _Float16* __restrict__ Bp,
    const float* __restrict__ bias, int M, int N, int K,
    _Float16* __restrict__ D)
{
  const int colGroups = N >> 6;
  const int waveInBlk = threadIdx.x >> 5;
  const int rowTile = (blockIdx.x / colGroups) * 8 + waveInBlk;
  const int cg = blockIdx.x % colGroups;
  const int m0 = rowTile << 4;
  if (m0 >= M) return;
  const int lane = threadIdx.x & 31;
  const int l15  = lane & 15;
  const int hi   = lane >> 4;
  const int ksteps = K >> 5;

  v8f acc[4];
#pragma unroll
  for (int t = 0; t < 4; ++t) {
    const float bv = bias[(cg << 6) + (t << 4) + l15];
#pragma unroll
    for (int r = 0; r < 8; ++r) acc[t][r] = bv;
  }

  const AT* Ap = A + (size_t)(m0 + l15) * K;
  // per-lane base into packed B for this column group
  const _Float16* BpBase = Bp + ((size_t)(cg << 2) * ksteps * 32 + lane) * 16;

  for (int s = 0; s < ksteps; ++s) {
    const int k0 = s << 5;
    // Issue every load for this K-step first (A: 2-4 x b128, B: 8 x b128).
    const h8 alo = loadA8(Ap + k0 + hi * 16);
    const h8 ahi = loadA8(Ap + k0 + hi * 16 + 8);
    h8 b0[4], b1[4];
#pragma unroll
    for (int t = 0; t < 4; ++t) {
      const h8* bp = (const h8*)(BpBase + (size_t)(t * ksteps + s) * 32 * 16);
      b0[t] = bp[0];
      b1[t] = bp[1];
    }
    v16h a;
#pragma unroll
    for (int j = 0; j < 8; ++j) { a[j] = alo[j]; a[8 + j] = ahi[j]; }
#pragma unroll
    for (int t = 0; t < 4; ++t) {
      v16h b;
#pragma unroll
      for (int j = 0; j < 8; ++j) { b[j] = b0[t][j]; b[8 + j] = b1[t][j]; }
      acc[t] = __builtin_amdgcn_wmma_f32_16x16x32_f16(false, a, false, b, (short)0,
                                                      acc[t], false, false);
    }
  }

#pragma unroll
  for (int t = 0; t < 4; ++t) {
#pragma unroll
    for (int r = 0; r < 8; ++r) {
      const int row = m0 + hi * 8 + r;   // C/D: VGPR r -> M = r + 8*hi
      float v = acc[t][r];
      if (RELU) v = fmaxf(v, 0.0f);
      D[(size_t)row * N + (cg << 6) + (t << 4) + l15] = (_Float16)v;
    }
  }
}

// ---------- edge pass 1: logits + segment max ----------
// One wave per edge; lane covers 8 channels; 8-lane groups = one head.
__global__ __launch_bounds__(256) void edge_logits_kernel(
    const _Float16* __restrict__ xl, const _Float16* __restrict__ xr,
    const float* __restrict__ ea, const float* __restrict__ We /*[16,256]*/,
    const float* __restrict__ att /*[4*64] flat*/, const int* __restrict__ src,
    const int* __restrict__ dst, float* __restrict__ logit,
    unsigned* __restrict__ mEnc, int E)
{
  const int e = (blockIdx.x * blockDim.x + threadIdx.x) >> 5;
  if (e >= E) return;
  const int lane = threadIdx.x & 31;
  const int s = src[e], d = dst[e];
  float eav[16];
#pragma unroll
  for (int k = 0; k < 16; ++k) eav[k] = ea[(size_t)e * 16 + k];
  const int ch0  = lane * 8;
  const int head = lane >> 3;
  const h8 xls = *(const h8*)(xl + (size_t)s * 256 + ch0);
  const h8 xrd = *(const h8*)(xr + (size_t)d * 256 + ch0);
  float lsum = 0.0f;
#pragma unroll
  for (int j = 0; j < 8; ++j) {
    const int ch = ch0 + j;
    float em = 0.0f;
#pragma unroll
    for (int k = 0; k < 16; ++k) em = fmaf(eav[k], We[k * 256 + ch], em);
    float z = (float)xls[j] + (float)xrd[j] + em;
    z = (z > 0.0f) ? z : 0.2f * z;              // leaky_relu 0.2
    lsum = fmaf(z, att[ch], lsum);              // att[h*64+c] == att[ch]
  }
  lsum += __shfl_xor(lsum, 1, 32);
  lsum += __shfl_xor(lsum, 2, 32);
  lsum += __shfl_xor(lsum, 4, 32);
  if ((lane & 7) == 0) {
    logit[(size_t)e * 4 + head] = lsum;
    atomicMax(&mEnc[d * 4 + head], encMax(lsum));
  }
}

// ---------- edge pass 2: exp + denominator + unnormalized aggregation ----------
__global__ __launch_bounds__(256) void edge_aggregate_kernel(
    const _Float16* __restrict__ xl, const float* __restrict__ logit,
    const unsigned* __restrict__ mEnc, const int* __restrict__ src,
    const int* __restrict__ dst, float* __restrict__ sden,
    float* __restrict__ out_un, int E)
{
  const int e = (blockIdx.x * blockDim.x + threadIdx.x) >> 5;
  if (e >= E) return;
  const int lane = threadIdx.x & 31;
  const int s = src[e], d = dst[e];
  const int head = lane >> 3, ch0 = lane * 8;
  const float m  = decMax(mEnc[d * 4 + head]);
  const float ev = __expf(logit[(size_t)e * 4 + head] - m);
  if ((lane & 7) == 0) atomAddF(&sden[d * 4 + head], ev);
  const h8 xls = *(const h8*)(xl + (size_t)s * 256 + ch0);
#pragma unroll
  for (int j = 0; j < 8; ++j) {
    atomAddF(&out_un[(size_t)d * 256 + ch0 + j], ev * (float)xls[j]);
  }
}

// ---------- node finalize layer0: normalize + bias (in place) ----------
__global__ __launch_bounds__(256) void gat_final_l0_kernel(
    float* __restrict__ out_un, const float* __restrict__ sden,
    const float* __restrict__ bias, int total)
{
  int i = blockIdx.x * blockDim.x + threadIdx.x;
  if (i >= total) return;
  const int n = i >> 8, ch = i & 255, head = ch >> 6;
  out_un[i] = out_un[i] / (sden[n * 4 + head] + 1e-16f) + bias[ch];
}

// ---------- node finalize layer1: normalize + mean over heads + bias ----------
__global__ __launch_bounds__(256) void gat_final_l1_kernel(
    const float* __restrict__ out_un, const float* __restrict__ sden,
    const float* __restrict__ bias, float* __restrict__ h2, int total)
{
  int i = blockIdx.x * blockDim.x + threadIdx.x;
  if (i >= total) return;
  const int n = i >> 6, c = i & 63;
  float acc = 0.0f;
#pragma unroll
  for (int h = 0; h < 4; ++h)
    acc += out_un[(size_t)n * 256 + h * 64 + c] / (sden[n * 4 + h] + 1e-16f);
  h2[i] = 0.25f * acc + bias[c];
}

// ---------- batchnorm stats: per-channel sum / sumsq ----------
template <int C>
__global__ __launch_bounds__(256) void bn_stats_kernel(
    const float* __restrict__ y, float* __restrict__ chSum,
    float* __restrict__ chSq, int Nn)
{
  const int ch = threadIdx.x % C;
  const int ro = threadIdx.x / C;
  const int RPB = 256 / C;
  float s = 0.0f, q = 0.0f;
  for (int n = blockIdx.x * RPB + ro; n < Nn; n += gridDim.x * RPB) {
    float v = y[(size_t)n * C + ch];
    s += v; q += v * v;
  }
  atomAddF(&chSum[ch], s);
  atomAddF(&chSq[ch], q);
}

__global__ __launch_bounds__(256) void bn_params_kernel(
    const float* chSum, const float* chSq, const float* g, const float* beta,
    float* scale, float* shift, int C, float invN)
{
  int c = blockIdx.x * blockDim.x + threadIdx.x;
  if (c >= C) return;
  float mu  = chSum[c] * invN;
  float var = chSq[c] * invN - mu * mu;
  float sc  = g[c] * rsqrtf(var + 1e-5f);
  scale[c] = sc;
  shift[c] = beta[c] - mu * sc;
}

__global__ __launch_bounds__(256) void bn_apply_relu_f16_kernel(
    const float* __restrict__ y, const float* __restrict__ scale,
    const float* __restrict__ shift, _Float16* __restrict__ dst, int total, int C)
{
  int i = blockIdx.x * blockDim.x + threadIdx.x;
  if (i >= total) return;
  const int ch = i % C;
  float v = fmaf(y[i], scale[ch], shift[ch]);
  dst[i] = (_Float16)fmaxf(v, 0.0f);
}

__global__ __launch_bounds__(256) void bn_apply_f32_kernel(
    float* __restrict__ y, const float* __restrict__ scale,
    const float* __restrict__ shift, int total, int C)
{
  int i = blockIdx.x * blockDim.x + threadIdx.x;
  if (i >= total) return;
  const int ch = i % C;
  y[i] = fmaf(y[i], scale[ch], shift[ch]);
}

// ---------- global mean pool (accumulate) ----------
__global__ __launch_bounds__(256) void pool_kernel(
    const float* __restrict__ h2, const int* __restrict__ batch,
    float* __restrict__ pooled, float* __restrict__ cnt, int total)
{
  int i = blockIdx.x * blockDim.x + threadIdx.x;
  if (i >= total) return;
  const int n = i >> 6, c = i & 63;
  const int g = batch[n];
  atomAddF(&pooled[g * 64 + c], h2[i]);
  if (c == 0) atomAddF(&cnt[g], 1.0f);
}

// ---------- MLP head: one graph per thread ----------
__global__ __launch_bounds__(256) void mlp_kernel(
    const float* __restrict__ pooled, const float* __restrict__ cnt,
    const float* __restrict__ Wm1, const float* __restrict__ bm1,
    const float* __restrict__ Wm2, const float* __restrict__ bm2,
    const float* __restrict__ Wm3, const float* __restrict__ bm3,
    float* __restrict__ out)
{
  int g = threadIdx.x;
  if (g >= NGRAPHS) return;
  const float inv = 1.0f / fmaxf(cnt[g], 1.0f);
  float p[64];
  for (int i = 0; i < 64; ++i) p[i] = pooled[g * 64 + i] * inv;
  float z1[32];
  for (int j = 0; j < 32; ++j) {
    float s = bm1[j];
    for (int i = 0; i < 64; ++i) s = fmaf(p[i], Wm1[i * 32 + j], s);
    z1[j] = fmaxf(s, 0.0f);
  }
  float z2[16];
  for (int j = 0; j < 16; ++j) {
    float s = bm2[j];
    for (int i = 0; i < 32; ++i) s = fmaf(z1[i], Wm2[i * 16 + j], s);
    z2[j] = fmaxf(s, 0.0f);
  }
  float o = bm3[0];
  for (int i = 0; i < 16; ++i) o = fmaf(z2[i], Wm3[i], o);
  out[g] = o;
}

// ---------- host-side launchers ----------

static inline void fillU32(hipStream_t s, void* p, unsigned v, int n) {
  fill_u32_kernel<<<(n + 255) / 256, 256, 0, s>>>((unsigned*)p, v, n);
}

static inline void packB(hipStream_t s, const float* B, _Float16* Bp, int K, int N) {
  pack_b_kernel<<<(K * N + 255) / 256, 256, 0, s>>>(B, Bp, K, N);
}

template <typename AT>
static void gemmLaunch(hipStream_t s, const AT* A, const _Float16* Bp,
                       const float* bias, int M, int N, int K,
                       _Float16* D, bool relu) {
  const int rowTiles = (M + 15) / 16, colGroups = N / 64;
  const int blocks = colGroups * ((rowTiles + 7) / 8);
  if (relu)
    wmma_gemm_kernel<AT, true><<<blocks, 256, 0, s>>>(A, Bp, bias, M, N, K, D);
  else
    wmma_gemm_kernel<AT, false><<<blocks, 256, 0, s>>>(A, Bp, bias, M, N, K, D);
}

extern "C" void kernel_launch(void* const* d_in, const int* in_sizes, int n_in,
                              void* d_out, int out_size, void* d_ws, size_t ws_size,
                              hipStream_t stream) {
  (void)in_sizes; (void)n_in; (void)out_size; (void)ws_size;
  const float* x         = (const float*)d_in[0];
  const float* edge_attr = (const float*)d_in[1];
  const int*   edge_idx  = (const int*)d_in[2];
  const int*   batch     = (const int*)d_in[3];
  const float* W_in = (const float*)d_in[4];  const float* b_in = (const float*)d_in[5];
  const float* Wl0  = (const float*)d_in[6];  const float* bl0  = (const float*)d_in[7];
  const float* Wr0  = (const float*)d_in[8];  const float* br0  = (const float*)d_in[9];
  const float* We0  = (const float*)d_in[10]; const float* att0 = (const float*)d_in[11];
  const float* bias0= (const float*)d_in[12]; const float* g0   = (const float*)d_in[13];
  const float* beta0= (const float*)d_in[14];
  const float* Wl1  = (const float*)d_in[15]; const float* bl1  = (const float*)d_in[16];
  const float* Wr1  = (const float*)d_in[17]; const float* br1  = (const float*)d_in[18];
  const float* We1  = (const float*)d_in[19]; const float* att1 = (const float*)d_in[20];
  const float* bias1= (const float*)d_in[21]; const float* g1   = (const float*)d_in[22];
  const float* beta1= (const float*)d_in[23];
  const float* Wm1  = (const float*)d_in[24]; const float* bm1  = (const float*)d_in[25];
  const float* Wm2  = (const float*)d_in[26]; const float* bm2  = (const float*)d_in[27];
  const float* Wm3  = (const float*)d_in[28]; const float* bm3  = (const float*)d_in[29];
  const int* src = edge_idx;
  const int* dst = edge_idx + NEDGES;

  // workspace carve-up (256B aligned)
  char* ws = (char*)d_ws;
  size_t off = 0;
  auto carve = [&](size_t bytes) -> void* {
    void* p = ws + off;
    off += (bytes + 255) & ~(size_t)255;
    return p;
  };
  _Float16* h0h    = (_Float16*)carve((size_t)NNODES * 64 * 2);
  _Float16* xlh    = (_Float16*)carve((size_t)NNODES * 256 * 2);
  _Float16* xrh    = (_Float16*)carve((size_t)NNODES * 256 * 2);
  float*    out_un = (float*)   carve((size_t)NNODES * 256 * 4);
  _Float16* h1h    = (_Float16*)carve((size_t)NNODES * 256 * 2);
  float*    h2     = (float*)   carve((size_t)NNODES * 64 * 4);
  float*    logit  = (float*)   carve((size_t)NEDGES * 4 * 4);
  unsigned* mEnc   = (unsigned*)carve((size_t)NNODES * 4 * 4);
  float*    sden   = (float*)   carve((size_t)NNODES * 4 * 4);
  float*    chSum  = (float*)   carve(256 * 4);
  float*    chSq   = (float*)   carve(256 * 4);
  float*    scale  = (float*)   carve(256 * 4);
  float*    shift  = (float*)   carve(256 * 4);
  float*    pooled = (float*)   carve(NGRAPHS * 64 * 4);
  float*    cnt    = (float*)   carve(NGRAPHS * 4);
  _Float16* WinP   = (_Float16*)carve(64 * 64 * 2);
  _Float16* Wl0P   = (_Float16*)carve(64 * 256 * 2);
  _Float16* Wr0P   = (_Float16*)carve(64 * 256 * 2);
  _Float16* Wl1P   = (_Float16*)carve(256 * 256 * 2);
  _Float16* Wr1P   = (_Float16*)carve(256 * 256 * 2);

  const unsigned NEG_INF_ENC = 0x007FFFFFu; // encMax(-inf)
  const int edgeBlocks = (NEDGES + 7) / 8;  // one wave per edge, 8 waves/block

  // weight packing into WMMA B-fragment order (f16)
  packB(stream, W_in, WinP, 64, 64);
  packB(stream, Wl0, Wl0P, 64, 256);
  packB(stream, Wr0, Wr0P, 64, 256);
  packB(stream, Wl1, Wl1P, 256, 256);
  packB(stream, Wr1, Wr1P, 256, 256);

  // ===== input projection: h0 = relu(x @ W_in + b_in), stored f16 =====
  gemmLaunch<float>(stream, x, WinP, b_in, NNODES, 64, 64, h0h, true);

  // ===== GATv2 layer 0 =====
  fillU32(stream, mEnc, NEG_INF_ENC, NNODES * 4);
  fillU32(stream, sden, 0, NNODES * 4);
  fillU32(stream, out_un, 0, NNODES * 256);
  fillU32(stream, chSum, 0, 256);
  fillU32(stream, chSq, 0, 256);

  gemmLaunch<_Float16>(stream, h0h, Wl0P, bl0, NNODES, 256, 64, xlh, false);
  gemmLaunch<_Float16>(stream, h0h, Wr0P, br0, NNODES, 256, 64, xrh, false);

  edge_logits_kernel<<<edgeBlocks, 256, 0, stream>>>(
      xlh, xrh, edge_attr, We0, att0, src, dst, logit, mEnc, NEDGES);
  edge_aggregate_kernel<<<edgeBlocks, 256, 0, stream>>>(
      xlh, logit, mEnc, src, dst, sden, out_un, NEDGES);

  gat_final_l0_kernel<<<(NNODES * 256) / 256, 256, 0, stream>>>(
      out_un, sden, bias0, NNODES * 256);

  bn_stats_kernel<256><<<512, 256, 0, stream>>>(out_un, chSum, chSq, NNODES);
  bn_params_kernel<<<1, 256, 0, stream>>>(chSum, chSq, g0, beta0, scale, shift,
                                          256, 1.0f / NNODES);
  bn_apply_relu_f16_kernel<<<(NNODES * 256) / 256, 256, 0, stream>>>(
      out_un, scale, shift, h1h, NNODES * 256, 256);

  // ===== GATv2 layer 1 =====
  fillU32(stream, mEnc, NEG_INF_ENC, NNODES * 4);
  fillU32(stream, sden, 0, NNODES * 4);
  fillU32(stream, out_un, 0, NNODES * 256);
  fillU32(stream, chSum, 0, 256);
  fillU32(stream, chSq, 0, 256);
  fillU32(stream, pooled, 0, NGRAPHS * 64);
  fillU32(stream, cnt, 0, NGRAPHS);

  gemmLaunch<_Float16>(stream, h1h, Wl1P, bl1, NNODES, 256, 256, xlh, false);
  gemmLaunch<_Float16>(stream, h1h, Wr1P, br1, NNODES, 256, 256, xrh, false);

  edge_logits_kernel<<<edgeBlocks, 256, 0, stream>>>(
      xlh, xrh, edge_attr, We1, att1, src, dst, logit, mEnc, NEDGES);
  edge_aggregate_kernel<<<edgeBlocks, 256, 0, stream>>>(
      xlh, logit, mEnc, src, dst, sden, out_un, NEDGES);

  gat_final_l1_kernel<<<(NNODES * 64) / 256, 256, 0, stream>>>(
      out_un, sden, bias1, h2, NNODES * 64);

  bn_stats_kernel<64><<<512, 256, 0, stream>>>(h2, chSum, chSq, NNODES);
  bn_params_kernel<<<1, 256, 0, stream>>>(chSum, chSq, g1, beta1, scale, shift,
                                          64, 1.0f / NNODES);
  bn_apply_f32_kernel<<<(NNODES * 64) / 256, 256, 0, stream>>>(
      h2, scale, shift, NNODES * 64, 64);

  // ===== pool + MLP head =====
  pool_kernel<<<(NNODES * 64) / 256, 256, 0, stream>>>(h2, batch, pooled, cnt,
                                                       NNODES * 64);
  mlp_kernel<<<1, 256, 0, stream>>>(pooled, cnt, Wm1, bm1, Wm2, bm2, Wm3, bm3,
                                    (float*)d_out);
}
